// PassModel_GIN_52785148068157
// MI455X (gfx1250) — compile-verified, matrix operand order checked by reading, not run on hardware
//
#include <hip/hip_runtime.h>
#include <cstdint>
#include <cstddef>

#define N_NODES 40000
#define T_STEPS 50
#define N_EDGES 500000
#define N_QUERY 100000
#define HDIM    48
#define DDIM    144
#define PHID    1024

typedef __attribute__((ext_vector_type(16))) _Float16     v16h;
typedef __attribute__((ext_vector_type(8)))  float        v8f;
typedef __attribute__((ext_vector_type(4)))  unsigned int u32x4;

union AF { v16h v; unsigned int u[8]; _Float16 h[16]; };

typedef __attribute__((address_space(1))) const u32x4 gu4_t;
typedef __attribute__((address_space(1))) const char  gchar_t;

__device__ __forceinline__ float sigmf(float x) { return 1.0f / (1.0f + __expf(-x)); }
// tanh(x) = 1 - 2/(exp(2x)+1): short VALU chain instead of the libm expansion.
__device__ __forceinline__ float tanhfast(float x) {
  float e = __expf(2.0f * x);
  return 1.0f - 2.0f / (e + 1.0f);
}

__device__ __forceinline__ v8f wmma_f16(v16h a, v16h b, v8f c) {
  return __builtin_amdgcn_wmma_f32_16x16x32_f16(false, a, false, b, (short)0, c, false, false);
}

// Per-lane byte offset into a [n][kpad] f16 weight: row select + 16B half-select.
__device__ __forceinline__ unsigned boff(int kpad, int lane) {
  return (unsigned)((lane & 15) * kpad * 2 + ((lane >= 16) ? 16 : 0));
}

// B fragment (K x N, f16): SGPR base + 32-bit lane offset + constant displacement
// -> global_load_b128 saddr-form with immediate offsets.
__device__ __forceinline__ v16h load_bfrag(const _Float16* __restrict__ W, unsigned off0,
                                           unsigned disp) {
  gchar_t* base = (gchar_t*)(unsigned long long)W;
  u32x4 lo = *(const gu4_t*)(base + (off0 + disp));
  u32x4 hi = *(const gu4_t*)(base + (off0 + disp + 32));
  AF f;
  f.u[0] = lo[0]; f.u[1] = lo[1]; f.u[2] = lo[2]; f.u[3] = lo[3];
  f.u[4] = hi[0]; f.u[5] = hi[1]; f.u[6] = hi[2]; f.u[7] = hi[3];
  return f.v;
}

// A fragment (M x 32, f16) from an LDS tile stored row-major [16][stride].
__device__ __forceinline__ v16h load_afrag_lds(const _Float16* S, int stride, int kc, int lane) {
  const u32x4* p = (const u32x4*)(S + (size_t)(lane & 15) * stride + kc);
  const int s0 = (lane >= 16) ? 1 : 0;
  u32x4 lo = p[s0];
  u32x4 hi = p[2 + s0];
  AF f;
  f.u[0] = lo[0]; f.u[1] = lo[1]; f.u[2] = lo[2]; f.u[3] = lo[3];
  f.u[4] = hi[0]; f.u[5] = hi[1]; f.u[6] = hi[2]; f.u[7] = hi[3];
  return f.v;
}

// -------- weight prep: f32 [rows][K] -> f16 [rows][kpad]; bias folded into pad row -----
__global__ void prep_w_kernel(const float* __restrict__ in, const float* __restrict__ bias,
                              _Float16* __restrict__ out, int rows, int K, int kpad,
                              int bias_k) {
  long i = (long)blockIdx.x * blockDim.x + threadIdx.x;
  long tot = (long)rows * kpad;
  if (i >= tot) return;
  int r = (int)(i / kpad), k = (int)(i % kpad);
  float v = 0.0f;
  if (k < K)            v = in[(long)r * K + k];
  else if (k == bias_k) v = bias[r];
  out[i] = (_Float16)v;
}

__global__ void zero_kernel(float* __restrict__ p, long n) {
  long i = (long)blockIdx.x * blockDim.x + threadIdx.x;
  if (i < n) p[i] = 0.0f;
}

// ---------------- GRU: one wave owns a 16-node tile; 3 layers pipelined per timestep ----
// Hidden state lives ONLY in the LDS stage tile (f16). Column 48 of every stage row is a
// persistent 1.0 so the padded weight row k=48 (holding the bias) is added by the WMMA.
__device__ __forceinline__ void gru_layer_step(
    const v16h* ain, int nchunks,
    const _Float16* __restrict__ Wih, int kpad_in,
    const _Float16* __restrict__ Whh,
    _Float16* hstage, int lane) {
  // A fragments of h_{t-1} (before we overwrite the stage)
  v16h ah0 = load_afrag_lds(hstage, 64, 0, lane);
  v16h ah1 = load_afrag_lds(hstage, 64, 32, lane);
  const int col = lane & 15;
  const int mb = (lane >= 16) ? 8 : 0;
  const unsigned offW = boff(kpad_in, lane);
  const unsigned offU = boff(64, lane);
  const v8f vzero = {};
  v8f acc[9], sep[3];
#pragma unroll
  for (int j = 0; j < 9; ++j) acc[j] = vzero;
#pragma unroll
  for (int j = 0; j < 3; ++j) sep[j] = vzero;
  // gi = x @ Wih^T (+ bih via pad row)
#pragma unroll
  for (int j = 0; j < 9; ++j) {
#pragma unroll
    for (int c = 0; c < 2; ++c)
      if (c < nchunks)
        acc[j] = wmma_f16(ain[c],
                          load_bfrag(Wih, offW, (unsigned)((16 * j * kpad_in + 32 * c) * 2)),
                          acc[j]);
  }
  // gh = h @ Whh^T (+ bhh via pad row); n-gate part kept separate (r * gh_n)
#pragma unroll
  for (int j = 0; j < 9; ++j) {
    v8f t = (j < 6) ? acc[j] : sep[j - 6];
    t = wmma_f16(ah0, load_bfrag(Whh, offU, (unsigned)((16 * j * 64) * 2)), t);
    t = wmma_f16(ah1, load_bfrag(Whh, offU, (unsigned)((16 * j * 64 + 32) * 2)), t);
    if (j < 6) acc[j] = t; else sep[j - 6] = t;
  }
#pragma unroll
  for (int j = 0; j < 3; ++j) {
#pragma unroll
    for (int r = 0; r < 8; ++r) {
      float hprev = (float)hstage[(r + mb) * 64 + 16 * j + col];
      float rr = sigmf(acc[j][r]);
      float zz = sigmf(acc[j + 3][r]);
      float nn = tanhfast(acc[j + 6][r] + rr * sep[j][r]);
      float hn = (1.0f - zz) * nn + zz * hprev;
      hstage[(r + mb) * 64 + 16 * j + col] = (_Float16)hn;
    }
  }
}

__global__ __launch_bounds__(32) void gru3_kernel(
    const float* __restrict__ x,
    const _Float16* __restrict__ wih0, const _Float16* __restrict__ wih1,
    const _Float16* __restrict__ wih2,
    const _Float16* __restrict__ whh0, const _Float16* __restrict__ whh1,
    const _Float16* __restrict__ whh2,
    float* __restrict__ out_all,    // [N][144] concat of 3 layer hT, or null
    float* __restrict__ out_last) { // [N][48] last layer hT, or null
  __shared__ alignas(16) _Float16 hst[3][16 * 64];
  const int lane = threadIdx.x;
  const int nb = blockIdx.x * 16;
  // h0 = 0; pad col 48 holds the constant 1.0 that activates the bias row of the weights.
  for (int i = lane; i < 3 * 16 * 64; i += 32)
    (&hst[0][0])[i] = ((i & 63) == 48) ? (_Float16)1.0f : (_Float16)0.0f;
  __syncthreads();

  const int m = lane & 15;
  const int hi = (lane >= 16) ? 8 : 0;
#pragma unroll 1
  for (int t = 0; t < T_STEPS; ++t) {
    // Launder weight pointers so the (loop-invariant) fragment loads are not hoisted
    // out of the t-loop and spilled to scratch.
    const _Float16* w0 = wih0; const _Float16* w1 = wih1; const _Float16* w2 = wih2;
    const _Float16* u0 = whh0; const _Float16* u1 = whh1; const _Float16* u2 = whh2;
    asm volatile("" : "+s"(w0), "+s"(w1), "+s"(w2), "+s"(u0), "+s"(u1), "+s"(u2));

    // layer 0 input: x[nb+m][t][0..2], k=3 -> 1.0 (bias row), rest zero-padded
    AF ax;
#pragma unroll
    for (int i = 0; i < 16; ++i) {
      int v = i >> 1, p = i & 1;
      int k = ((v < 4) ? 0 : 16) + hi + 2 * (v & 3) + p;
      ax.h[i] = (k < 3) ? (_Float16)x[((size_t)(nb + m) * T_STEPS + t) * 3 + k]
                        : ((k == 3) ? (_Float16)1.0f : (_Float16)0.0f);
    }
    v16h a0[2]; a0[0] = ax.v;
    gru_layer_step(a0, 1, w0, 32, u0, hst[0], lane);
    v16h a1[2] = { load_afrag_lds(hst[0], 64, 0, lane), load_afrag_lds(hst[0], 64, 32, lane) };
    gru_layer_step(a1, 2, w1, 64, u1, hst[1], lane);
    v16h a2[2] = { load_afrag_lds(hst[1], 64, 0, lane), load_afrag_lds(hst[1], 64, 32, lane) };
    gru_layer_step(a2, 2, w2, 64, u2, hst[2], lane);
  }
  const int col = lane & 15, mb = hi;
  if (out_all) {
#pragma unroll
    for (int l = 0; l < 3; ++l)
#pragma unroll
      for (int j = 0; j < 3; ++j)
#pragma unroll
        for (int r = 0; r < 8; ++r)
          out_all[(size_t)(nb + r + mb) * DDIM + 48 * l + 16 * j + col] =
              (float)hst[l][(r + mb) * 64 + 16 * j + col];
  }
  if (out_last) {
#pragma unroll
    for (int j = 0; j < 3; ++j)
#pragma unroll
      for (int r = 0; r < 8; ++r)
        out_last[(size_t)(nb + r + mb) * HDIM + 16 * j + col] =
            (float)hst[2][(r + mb) * 64 + 16 * j + col];
  }
}

// ---------------- GIN: per-node attention scores ---------------------------------------
__global__ void gin_scores_kernel(const float* __restrict__ h, const float* __restrict__ attW,
                                  float* __restrict__ s, float* __restrict__ t, int n) {
  int node = (int)((blockIdx.x * blockDim.x + threadIdx.x) >> 5);
  int lane = threadIdx.x & 31;
  if (node >= n) return;
  const float* row = h + (size_t)node * DDIM;
  float ss = 0.0f, tt = 0.0f;
  for (int c = lane; c < DDIM; c += 32) {
    float v = row[c];
    ss += v * attW[c];
    tt += v * attW[DDIM + c];
  }
#pragma unroll
  for (int msk = 16; msk > 0; msk >>= 1) {
    ss += __shfl_xor(ss, msk, 32);
    tt += __shfl_xor(tt, msk, 32);
  }
  if (lane == 0) { s[node] = ss; t[node] = tt; }
}

// ---------------- GIN: edge scatter (one wave per edge, channels across lanes) ---------
__global__ void gin_edge_kernel(const float* __restrict__ h, const int* __restrict__ src,
                                const int* __restrict__ dst, const float* __restrict__ s,
                                const float* __restrict__ t, float* __restrict__ f, int ne) {
  int e = (int)((blockIdx.x * blockDim.x + threadIdx.x) >> 5);
  int lane = threadIdx.x & 31;
  if (e >= ne) return;
  int u = src[e], v = dst[e];
  float w = sigmf(s[u] + t[v]);
  const float* hr = h + (size_t)u * DDIM;
  float* fr = f + (size_t)v * DDIM;
  for (int c = lane; c < DDIM; c += 32)
    __hip_atomic_fetch_add(&fr[c], w * hr[c], __ATOMIC_RELAXED, __HIP_MEMORY_SCOPE_AGENT);
}

// ---------------- GIN: fc ((1+eps)h + f) @ W^T (+ ELU) with WMMA -----------------------
__global__ __launch_bounds__(32) void gin_fc_kernel(
    const float* __restrict__ h, const float* __restrict__ f,
    const _Float16* __restrict__ Wt, const float* __restrict__ eps,
    float* __restrict__ out, int do_elu) {
  __shared__ alignas(16) _Float16 As[16 * 160];
  const int lane = threadIdx.x;
  const int nb = blockIdx.x * 16;
  float ep = 1.0f + eps[0];
  for (int i = lane; i < 16 * 160; i += 32) {
    int m = i / 160, k = i % 160;
    float v = (k < DDIM)
                  ? ep * h[(size_t)(nb + m) * DDIM + k] + f[(size_t)(nb + m) * DDIM + k]
                  : 0.0f;
    As[i] = (_Float16)v;
  }
  __syncthreads();
  v16h af[5];
#pragma unroll
  for (int c = 0; c < 5; ++c) af[c] = load_afrag_lds(As, 160, 32 * c, lane);
  const int col = lane & 15, mb = (lane >= 16) ? 8 : 0;
  const unsigned offW = boff(160, lane);
  const v8f vzero = {};
#pragma unroll
  for (int j = 0; j < 9; ++j) {
    v8f acc = vzero;
#pragma unroll
    for (int c = 0; c < 5; ++c)
      acc = wmma_f16(af[c], load_bfrag(Wt, offW, (unsigned)((16 * j * 160 + 32 * c) * 2)), acc);
#pragma unroll
    for (int r = 0; r < 8; ++r) {
      float v = acc[r];
      if (do_elu) v = (v > 0.0f) ? v : (__expf(v) - 1.0f);
      out[(size_t)(nb + r + mb) * DDIM + 16 * j + col] = v;
    }
  }
}

// ---------------- predictor: union[Q,384(+1)] @ W1^T -> relu -> dot W2 -> sigmoid ------
// pb1 folded into pW1 pad row k=384; union stage col 384 = 1.0.
__global__ __launch_bounds__(32) void predictor_kernel(
    const float* __restrict__ g, const float* __restrict__ t2,
    const int* __restrict__ qf, const int* __restrict__ qt,
    const _Float16* __restrict__ W1t,
    const float* __restrict__ W2, const float* __restrict__ b2,
    float* __restrict__ out) {
  __shared__ alignas(16) _Float16 As[16 * 416];
  const int lane = threadIdx.x;
  const int qb = blockIdx.x * 16;
  for (int i = lane; i < 16 * 416; i += 32) {
    int m = i / 416, c = i % 416;
    int row = qb + m;
    float v;
    if (c < 144)      v = g[(size_t)qf[row] * DDIM + c];
    else if (c < 288) v = g[(size_t)qt[row] * DDIM + (c - 144)];
    else if (c < 336) v = t2[(size_t)qf[row] * HDIM + (c - 288)];
    else if (c < 384) v = t2[(size_t)qt[row] * HDIM + (c - 336)];
    else              v = (c == 384) ? 1.0f : 0.0f;
    As[i] = (_Float16)v;
  }
  __syncthreads();
  v16h af[13];
#pragma unroll
  for (int c = 0; c < 13; ++c) af[c] = load_afrag_lds(As, 416, 32 * c, lane);
  const int col = lane & 15, mb = (lane >= 16) ? 8 : 0;
  const unsigned offW = boff(416, lane);
  const v8f vzero = {};
  float p[8];
#pragma unroll
  for (int r = 0; r < 8; ++r) p[r] = 0.0f;
#pragma unroll 1
  for (int j = 0; j < 64; ++j) {
    v8f acc = vzero;
#pragma unroll
    for (int c = 0; c < 13; ++c)
      acc = wmma_f16(af[c], load_bfrag(W1t, offW, (unsigned)(16 * 416 * 2) * j + 64u * c), acc);
    float w2 = W2[16 * j + col];
#pragma unroll
    for (int r = 0; r < 8; ++r) {
      float hv = (acc[r] > 0.0f) ? acc[r] : 0.0f;
      p[r] += hv * w2;
    }
  }
#pragma unroll
  for (int r = 0; r < 8; ++r) {
#pragma unroll
    for (int msk = 8; msk > 0; msk >>= 1) p[r] += __shfl_xor(p[r], msk, 16);
  }
  if (col == 0) {
    float bb2 = b2[0];
#pragma unroll
    for (int r = 0; r < 8; ++r) out[qb + r + mb] = sigmf(p[r] + bb2);
  }
}

// ---------------------------------------------------------------------------------------
extern "C" void kernel_launch(void* const* d_in, const int* in_sizes, int n_in,
                              void* d_out, int out_size, void* d_ws, size_t ws_size,
                              hipStream_t stream) {
  (void)in_sizes; (void)n_in; (void)out_size; (void)ws_size;
  const float* x       = (const float*)d_in[0];
  const int*   src     = (const int*)d_in[1];
  const int*   dst     = (const int*)d_in[2];
  const int*   q_from  = (const int*)d_in[3];
  const int*   q_to    = (const int*)d_in[4];
  const float* na_Wih0 = (const float*)d_in[5];
  const float* na_Wih12= (const float*)d_in[6];
  const float* na_Whh  = (const float*)d_in[7];
  const float* na_bih  = (const float*)d_in[8];
  const float* na_bhh  = (const float*)d_in[9];
  const float* ta_Wih0 = (const float*)d_in[10];
  const float* ta_Wih12= (const float*)d_in[11];
  const float* ta_Whh  = (const float*)d_in[12];
  const float* ta_bih  = (const float*)d_in[13];
  const float* ta_bhh  = (const float*)d_in[14];
  const float* fc1_W   = (const float*)d_in[15];
  const float* eps1    = (const float*)d_in[16];
  const float* att1_W  = (const float*)d_in[17];
  const float* fc2_W   = (const float*)d_in[18];
  const float* eps2    = (const float*)d_in[19];
  const float* att2_W  = (const float*)d_in[20];
  const float* pW1     = (const float*)d_in[21];
  const float* pb1     = (const float*)d_in[22];
  const float* pW2     = (const float*)d_in[23];
  const float* pb2     = (const float*)d_in[24];
  float* out = (float*)d_out;

  char* ws = (char*)d_ws;
  size_t off = 0;
  auto alloc = [&](size_t bytes) -> char* {
    char* p = ws + off;
    off = (off + bytes + 255) & ~(size_t)255;
    return p;
  };

  // f16 weight buffers, row-major [n][kpad], biases folded into pad rows
  _Float16* naWih0h = (_Float16*)alloc(144 * 32 * 2);
  _Float16* naWih1h = (_Float16*)alloc(144 * 64 * 2);
  _Float16* naWih2h = (_Float16*)alloc(144 * 64 * 2);
  _Float16* naWhh0h = (_Float16*)alloc(144 * 64 * 2);
  _Float16* naWhh1h = (_Float16*)alloc(144 * 64 * 2);
  _Float16* naWhh2h = (_Float16*)alloc(144 * 64 * 2);
  _Float16* taWih0h = (_Float16*)alloc(144 * 32 * 2);
  _Float16* taWih1h = (_Float16*)alloc(144 * 64 * 2);
  _Float16* taWih2h = (_Float16*)alloc(144 * 64 * 2);
  _Float16* taWhh0h = (_Float16*)alloc(144 * 64 * 2);
  _Float16* taWhh1h = (_Float16*)alloc(144 * 64 * 2);
  _Float16* taWhh2h = (_Float16*)alloc(144 * 64 * 2);
  _Float16* fc1h    = (_Float16*)alloc(144 * 160 * 2);
  _Float16* fc2h    = (_Float16*)alloc(144 * 160 * 2);
  _Float16* pW1h    = (_Float16*)alloc((size_t)PHID * 416 * 2);
  float* traj  = (float*)alloc((size_t)N_NODES * DDIM * 4);
  float* traj2 = (float*)alloc((size_t)N_NODES * HDIM * 4);
  float* g1    = (float*)alloc((size_t)N_NODES * DDIM * 4);
  float* g2    = (float*)alloc((size_t)N_NODES * DDIM * 4);
  float* fbuf  = (float*)alloc((size_t)N_NODES * DDIM * 4);
  float* sbuf  = (float*)alloc((size_t)N_NODES * 4);
  float* tbuf  = (float*)alloc((size_t)N_NODES * 4);

  auto prep = [&](const float* in, const float* bias, _Float16* o,
                  int rows, int K, int kpad, int bias_k) {
    long tot = (long)rows * kpad;
    unsigned blocks = (unsigned)((tot + 255) / 256);
    prep_w_kernel<<<blocks, 256, 0, stream>>>(in, bias, o, rows, K, kpad, bias_k);
  };
  prep(na_Wih0,             na_bih + 0,   naWih0h, 144, 3, 32, 3);
  prep(na_Wih12 + 0*144*48, na_bih + 144, naWih1h, 144, 48, 64, 48);
  prep(na_Wih12 + 1*144*48, na_bih + 288, naWih2h, 144, 48, 64, 48);
  prep(na_Whh + 0*144*48,   na_bhh + 0,   naWhh0h, 144, 48, 64, 48);
  prep(na_Whh + 1*144*48,   na_bhh + 144, naWhh1h, 144, 48, 64, 48);
  prep(na_Whh + 2*144*48,   na_bhh + 288, naWhh2h, 144, 48, 64, 48);
  prep(ta_Wih0,             ta_bih + 0,   taWih0h, 144, 3, 32, 3);
  prep(ta_Wih12 + 0*144*48, ta_bih + 144, taWih1h, 144, 48, 64, 48);
  prep(ta_Wih12 + 1*144*48, ta_bih + 288, taWih2h, 144, 48, 64, 48);
  prep(ta_Whh + 0*144*48,   ta_bhh + 0,   taWhh0h, 144, 48, 64, 48);
  prep(ta_Whh + 1*144*48,   ta_bhh + 144, taWhh1h, 144, 48, 64, 48);
  prep(ta_Whh + 2*144*48,   ta_bhh + 288, taWhh2h, 144, 48, 64, 48);
  prep(fc1_W, nullptr, fc1h, 144, 144, 160, -1);
  prep(fc2_W, nullptr, fc2h, 144, 144, 160, -1);
  prep(pW1,   pb1,     pW1h, PHID, 384, 416, 384);

  // GRUs (WMMA): na -> traj [N,144], ta -> traj2 [N,48]
  gru3_kernel<<<N_NODES / 16, 32, 0, stream>>>(x, naWih0h, naWih1h, naWih2h,
                                               naWhh0h, naWhh1h, naWhh2h,
                                               traj, nullptr);
  gru3_kernel<<<N_NODES / 16, 32, 0, stream>>>(x, taWih0h, taWih1h, taWih2h,
                                               taWhh0h, taWhh1h, taWhh2h,
                                               nullptr, traj2);

  // GIN layer 1 (+ELU)
  gin_scores_kernel<<<N_NODES / 8, 256, 0, stream>>>(traj, att1_W, sbuf, tbuf, N_NODES);
  zero_kernel<<<(unsigned)(((long)N_NODES * DDIM + 255) / 256), 256, 0, stream>>>(
      fbuf, (long)N_NODES * DDIM);
  gin_edge_kernel<<<N_EDGES / 8, 256, 0, stream>>>(traj, src, dst, sbuf, tbuf, fbuf, N_EDGES);
  gin_fc_kernel<<<N_NODES / 16, 32, 0, stream>>>(traj, fbuf, fc1h, eps1, g1, 1);

  // GIN layer 2
  gin_scores_kernel<<<N_NODES / 8, 256, 0, stream>>>(g1, att2_W, sbuf, tbuf, N_NODES);
  zero_kernel<<<(unsigned)(((long)N_NODES * DDIM + 255) / 256), 256, 0, stream>>>(
      fbuf, (long)N_NODES * DDIM);
  gin_edge_kernel<<<N_EDGES / 8, 256, 0, stream>>>(g1, src, dst, sbuf, tbuf, fbuf, N_EDGES);
  gin_fc_kernel<<<N_NODES / 16, 32, 0, stream>>>(g1, fbuf, fc2h, eps2, g2, 0);

  // Predictor MLP (WMMA)
  predictor_kernel<<<N_QUERY / 16, 32, 0, stream>>>(g2, traj2, q_from, q_to,
                                                    pW1h, pW2, pb2, out);
}
